// MultiheadAttention_44375602102761
// MI455X (gfx1250) — matrix operand order repaired and checked by reading, not compiled, not run on hardware
//
#include <hip/hip_runtime.h>
#include <hip/hip_bf16.h>

typedef __bf16 bf16;
typedef __attribute__((ext_vector_type(16))) __bf16 v16bf;
typedef __attribute__((ext_vector_type(8)))  float  v8f;

struct U2 { unsigned int x, y; };

union BF16Frag {
    v16bf v;
    unsigned int u[8];
};

union BF16Pair {
    unsigned int u;
    bf16 h[2];
};

__device__ __forceinline__ v8f v8f_zero() {
    v8f z;
#pragma unroll
    for (int i = 0; i < 8; ++i) z[i] = 0.0f;
    return z;
}

__device__ __forceinline__ v16bf ones_frag() {
    BF16Frag f;
#pragma unroll
    for (int i = 0; i < 8; ++i) f.u[i] = 0x3F803F80u;   // packed bf16 {1.0, 1.0}
    return f.v;
}

__device__ __forceinline__ v8f wmma_bf16(v16bf a, v16bf b, v8f c) {
    return __builtin_amdgcn_wmma_f32_16x16x32_bf16(
        /*neg_a=*/false, a, /*neg_b=*/false, b,
        /*c_mod=*/(short)0, c, /*reuse_a=*/false, /*reuse_b=*/false);
}

// CDNA5 async global->LDS copy (ASYNCcnt-tracked). The VDST operand is the
// 32-bit LDS byte offset; flat LDS pointers keep the offset in addr[31:0]
// (aperture tag lives in the high 32 bits), so truncation yields the offset.
__device__ __forceinline__ void async_copy_b128(unsigned int lds_off, const void* gsrc) {
    asm volatile("global_load_async_to_lds_b128 %0, %1, off"
                 :: "v"(lds_off), "v"((unsigned long long)(size_t)gsrc)
                 : "memory");
}

__device__ __forceinline__ void wait_asynccnt0() {
#if __has_builtin(__builtin_amdgcn_s_wait_asynccnt)
    __builtin_amdgcn_s_wait_asynccnt(0);
#else
    asm volatile("s_wait_asynccnt 0x0" ::: "memory");
#endif
}

// A-fragment (16x32, M x K, bf16). Lane m=lane&15 supplies row m; lanes 16-31
// hold the K=+8 halves. VGPR v(0..3): K = half*8 + 2v; v(4..7): K = 16+half*8+2(v-4).
__device__ __forceinline__ v16bf load_frag_a(const bf16* tile, int stride) {
    int lane = threadIdx.x & 31;
    int m = lane & 15, half = lane >> 4;
    const bf16* rp = tile + m * stride + half * 8;
    BF16Frag f;
    U2 t0 = *(const U2*)(rp);
    U2 t1 = *(const U2*)(rp + 4);
    U2 t2 = *(const U2*)(rp + 16);
    U2 t3 = *(const U2*)(rp + 20);
    f.u[0] = t0.x; f.u[1] = t0.y; f.u[2] = t1.x; f.u[3] = t1.y;
    f.u[4] = t2.x; f.u[5] = t2.y; f.u[6] = t3.x; f.u[7] = t3.y;
    return f.v;
}

// B-fragment (32x16, K x N, bf16) from K-contiguous (transposed) storage
// Bt[n][k]. VGPR v(0..3): K = half*8 + 2v; v(4..7): K = 16+half*8+2(v-4), column n = lane&15.
__device__ __forceinline__ v16bf load_frag_b(const bf16* tilet, int stride, int koff) {
    int lane = threadIdx.x & 31;
    int n = lane & 15, half = lane >> 4;
    const bf16* rp = tilet + n * stride + koff + half * 8;
    BF16Frag f;
    U2 t0 = *(const U2*)(rp);
    U2 t1 = *(const U2*)(rp + 4);
    U2 t2 = *(const U2*)(rp + 16);
    U2 t3 = *(const U2*)(rp + 20);
    f.u[0] = t0.x; f.u[1] = t0.y; f.u[2] = t1.x; f.u[3] = t1.y;
    f.u[4] = t2.x; f.u[5] = t2.y; f.u[6] = t3.x; f.u[7] = t3.y;
    return f.v;
}

// -------------------------------------------------------------------------
// Tiled GEMM: C[M,N] = ((A[M,Kd] * B[Kd,N]) + bias[N]) * cscale.
// bf16 WMMA, f32 accum. Block tile 128x128, 8 waves (4x2), wave tile 32x64.
// blockIdx.z selects head: B += z*strideB, bias += z*strideBias, C += z*strideC.
// f32->bf16 conversion fused into LDS staging.
// -------------------------------------------------------------------------
template <typename AT, bool OUT_BF16>
__global__ __launch_bounds__(256) void gemm_bf16_wmma(
    const AT* __restrict__ A, const float* __restrict__ B,
    const float* __restrict__ bias, void* __restrict__ Cout,
    int M, int N, int Kd, long strideB, long strideBias, long strideC,
    float cscale)
{
    __shared__ __align__(16) bf16 Al[128 * 40];   // A tile, row-major, pad 40
    __shared__ __align__(16) bf16 Bt[128 * 40];   // B tile, transposed [n][k]

    const float* Bp = B + (long)blockIdx.z * strideB;
    const float* biasp = bias ? (bias + (long)blockIdx.z * strideBias) : nullptr;

    const int m0 = blockIdx.x * 128;
    const int n0 = blockIdx.y * 128;
    const int tid = threadIdx.x;
    const int w = tid >> 5, lane = tid & 31;
    const int wm = w >> 1, wn = w & 1;
    const int half = lane >> 4, nl = lane & 15;

    v8f acc[2][4];
#pragma unroll
    for (int mi = 0; mi < 2; ++mi)
#pragma unroll
        for (int ni = 0; ni < 4; ++ni) acc[mi][ni] = v8f_zero();

    for (int k0 = 0; k0 < Kd; k0 += 32) {
        // Stage A tile (128x32) with f32->bf16 conversion.
#pragma unroll
        for (int i = 0; i < 16; ++i) {
            int idx = tid + i * 256;
            int r = idx >> 5, c = idx & 31;
            float v = (float)A[(long)(m0 + r) * Kd + (k0 + c)];
            Al[r * 40 + c] = (bf16)v;
        }
        // Stage B tile (32x128) transposed into Bt[n][k].
#pragma unroll
        for (int i = 0; i < 16; ++i) {
            int idx = tid + i * 256;
            int kk = idx >> 7, n = idx & 127;
            float v = Bp[(long)(k0 + kk) * N + (n0 + n)];
            Bt[n * 40 + kk] = (bf16)v;
        }
        __syncthreads();

        v16bf a0 = load_frag_a(Al + (wm * 32 + 0) * 40, 40);
        v16bf a1 = load_frag_a(Al + (wm * 32 + 16) * 40, 40);
        v16bf bfr[4];
#pragma unroll
        for (int ni = 0; ni < 4; ++ni)
            bfr[ni] = load_frag_b(Bt + (wn * 64 + ni * 16) * 40, 40, 0);

#pragma unroll
        for (int ni = 0; ni < 4; ++ni) {
            acc[0][ni] = wmma_bf16(a0, bfr[ni], acc[0][ni]);
            acc[1][ni] = wmma_bf16(a1, bfr[ni], acc[1][ni]);
        }

        __syncthreads();
    }

    // Epilogue: C/D layout row m = r + 8*half, col n = lane&15.
#pragma unroll
    for (int mi = 0; mi < 2; ++mi)
#pragma unroll
        for (int ni = 0; ni < 4; ++ni) {
            int col = n0 + wn * 64 + ni * 16 + nl;
            float bv = biasp ? biasp[col] : 0.0f;
#pragma unroll
            for (int r = 0; r < 8; ++r) {
                int row = m0 + wm * 32 + mi * 16 + r + half * 8;
                long off = (long)blockIdx.z * strideC + (long)row * N + col;
                float v = (acc[mi][ni][r] + bv) * cscale;
                if (OUT_BF16) ((bf16*)Cout)[off] = (bf16)v;
                else          ((float*)Cout)[off] = v;
            }
        }
}

// -------------------------------------------------------------------------
// Flash attention: grid (S/128, H), 256 threads = 8 waves.
// Wave w handles 16 query rows; streams keys in chunks of 32 with
// double-buffered LDS tiles. K tiles staged via CDNA5 async copies
// (global_load_async_to_lds_b128, ASYNCcnt); V staged manually (transposed).
// Q arrives pre-scaled by 1/sqrt(E). Row sums of P are accumulated on the
// matrix pipe via P x Ones (lands exactly in the C-layout of the l vector).
// -------------------------------------------------------------------------
__global__ __launch_bounds__(256) void attn_flash_wmma(
    const bf16* __restrict__ Qb, const bf16* __restrict__ Kb,
    const bf16* __restrict__ Vb, bf16* __restrict__ Zb)
{
    const int S = 2048, E = 128, HE = 2048;
    __shared__ __align__(16) bf16 Kl[2][32 * 136];   // 32 keys x 128 e (pad 136)
    __shared__ __align__(16) bf16 Vt[2][128 * 40];   // transposed: [e][key]
    __shared__ __align__(16) bf16 Ps[8][16 * 40];    // per-wave P scratch 16x32

    const int h = blockIdx.y;
    const int tid = threadIdx.x, w = tid >> 5, lane = tid & 31;
    const int half = lane >> 4, nl = lane & 15;
    const int q0 = blockIdx.x * 128 + w * 16;

    const bf16* Qh = Qb + (long)h * S * E;
    const bf16* Kh = Kb + (long)h * S * E;
    const bf16* Vh = Vb + (long)h * S * E;

    // Q fragments loaded directly from global (layout matches A-frag pattern).
    v16bf qf[4];
#pragma unroll
    for (int kk = 0; kk < 4; ++kk)
        qf[kk] = load_frag_a(Qh + (long)q0 * E + kk * 32, E);

    const v16bf onef = ones_frag();

    v8f O[8];
#pragma unroll
    for (int nt = 0; nt < 8; ++nt) O[nt] = v8f_zero();
    v8f Ls = v8f_zero();                 // running softmax denominator (per row)
    float mprev[8];
#pragma unroll
    for (int r = 0; r < 8; ++r) mprev[r] = -1e30f;

    // --- staging helpers (all 256 threads participate) ---
    auto stage_K_async = [&](int j0, bf16* dst) {
        unsigned int base = (unsigned int)(size_t)(void*)dst;  // LDS byte offset
#pragma unroll
        for (int i = 0; i < 2; ++i) {
            int c = tid + i * 256;               // 512 x 16B chunks = 8 KB
            int row = c >> 4, o16 = c & 15;      // 16 chunks per 256B key row
            const char* g = (const char*)(Kh + (long)(j0 + row) * E) + o16 * 16;
            async_copy_b128(base + row * 272 + o16 * 16, g);
        }
    };
    auto stage_V = [&](int j0, bf16* dst) {
#pragma unroll
        for (int i = 0; i < 8; ++i) {
            int idx = tid + i * 256;             // 2048 dwords = 4096 bf16
            int key = idx >> 6, e2 = idx & 63;   // e = 2*e2
            BF16Pair p;
            p.u = ((const unsigned int*)(Vh + (long)(j0 + key) * E))[e2];
            dst[(e2 * 2 + 0) * 40 + key] = p.h[0];
            dst[(e2 * 2 + 1) * 40 + key] = p.h[1];
        }
    };

    const int NJ = S / 32;
    stage_K_async(0, &Kl[0][0]);
    stage_V(0, &Vt[0][0]);
    wait_asynccnt0();
    __syncthreads();

    for (int jc = 0; jc < NJ; ++jc) {
        const int cur = jc & 1;
        if (jc + 1 < NJ) {                       // prefetch next tile pair
            stage_K_async((jc + 1) * 32, &Kl[cur ^ 1][0]);
            stage_V((jc + 1) * 32, &Vt[cur ^ 1][0]);
        }
        const bf16* Kc = &Kl[cur][0];
        const bf16* Vc = &Vt[cur][0];

        // Scores: S0 = Q*K(j..j+15)^T, S1 = Q*K(j+16..j+31)^T. Row-major K
        // rows in LDS are exactly transposed-B storage (Bt[n][e] = K[n][e]).
        v8f s0 = v8f_zero(), s1 = v8f_zero();
#pragma unroll
        for (int kk = 0; kk < 4; ++kk) {
            v16bf b0 = load_frag_b(Kc,            136, kk * 32);
            v16bf b1 = load_frag_b(Kc + 16 * 136, 136, kk * 32);
            s0 = wmma_bf16(qf[kk], b0, s0);
            s1 = wmma_bf16(qf[kk], b1, s1);
        }

        // Online softmax per row (row m = r + 8*half, cols across 16 lanes).
        // Only the max needs a cross-lane butterfly; row sums go to the
        // matrix pipe below.
        float alpha[8];
        bf16* Pw = Ps[w];
#pragma unroll
        for (int r = 0; r < 8; ++r) {
            float v0 = s0[r], v1 = s1[r];        // pre-scaled by 1/sqrt(E)
            float mx = fmaxf(v0, v1);
            mx = fmaxf(mx, __shfl_xor(mx, 1, 32));
            mx = fmaxf(mx, __shfl_xor(mx, 2, 32));
            mx = fmaxf(mx, __shfl_xor(mx, 4, 32));
            mx = fmaxf(mx, __shfl_xor(mx, 8, 32));
            float mn = fmaxf(mprev[r], mx);
            alpha[r] = __expf(mprev[r] - mn);
            mprev[r] = mn;
            int row = r + half * 8;
            Pw[row * 40 + nl]      = (bf16)__expf(v0 - mn);  // keys 0..15
            Pw[row * 40 + 16 + nl] = (bf16)__expf(v1 - mn);  // keys 16..31
        }

        // P (C-layout) -> A-fragment via per-wave LDS scratch.
        v16bf pf = load_frag_a(Pw, 40);

        // Denominator: Ls = alpha*Ls + rowsum(P) as P(16x32) x Ones(32x16).
        {
            v8f l = Ls;
#pragma unroll
            for (int r = 0; r < 8; ++r) l[r] *= alpha[r];
            Ls = wmma_bf16(pf, onef, l);
        }

        // O[nt] = alpha*O[nt] + P(16x32) * V(32x16 per e-tile).
#pragma unroll
        for (int nt = 0; nt < 8; ++nt) {
            v8f o = O[nt];
#pragma unroll
            for (int r = 0; r < 8; ++r) o[r] *= alpha[r];
            v16bf vb = load_frag_b(Vc + nt * 16 * 40, 40, 0);
            O[nt] = wmma_bf16(pf, vb, o);
        }

        wait_asynccnt0();     // own async copies into next buffer landed
        __syncthreads();      // everyone done reading cur / writing nxt
    }

    // Normalize and write Z (head-concatenated columns), bf16.
#pragma unroll
    for (int nt = 0; nt < 8; ++nt)
#pragma unroll
        for (int r = 0; r < 8; ++r) {
            int row = q0 + r + half * 8;
            int col = h * 128 + nt * 16 + nl;
            Zb[(long)row * HE + col] = (bf16)(O[nt][r] / Ls[r]);
        }
}

extern "C" void kernel_launch(void* const* d_in, const int* in_sizes, int n_in,
                              void* d_out, int out_size, void* d_ws, size_t ws_size,
                              hipStream_t stream) {
    (void)in_sizes; (void)n_in; (void)out_size; (void)ws_size;
    const int S = 2048, W = 1024, E = 128, H = 16;

    const float* x    = (const float*)d_in[0];
    const float* Wq   = (const float*)d_in[1];
    const float* bq   = (const float*)d_in[2];
    const float* Wk   = (const float*)d_in[3];
    const float* bk   = (const float*)d_in[4];
    const float* Wv   = (const float*)d_in[5];
    const float* bv   = (const float*)d_in[6];
    const float* proj = (const float*)d_in[7];

    const size_t perHSE = (size_t)H * S * E;       // 4M elements
    bf16* Qb = (bf16*)d_ws;
    bf16* Kb = Qb + perHSE;
    bf16* Vb = Kb + perHSE;
    bf16* Zb = Vb + perHSE;                        // [S][H*E]

    const float scale = 0.08838834764831845f;      // 1/sqrt(E), folded into Q

    dim3 blk(256);
    dim3 gqkv(S / 128, E / 128, H);                // (16, 1, 16)
    long sB = (long)W * E, sBias = E, sC = (long)S * E;

    gemm_bf16_wmma<float, true><<<gqkv, blk, 0, stream>>>(x, Wq, bq, Qb, S, E, W, sB, sBias, sC, scale);
    gemm_bf16_wmma<float, true><<<gqkv, blk, 0, stream>>>(x, Wk, bk, Kb, S, E, W, sB, sBias, sC, 1.0f);
    gemm_bf16_wmma<float, true><<<gqkv, blk, 0, stream>>>(x, Wv, bv, Vb, S, E, W, sB, sBias, sC, 1.0f);

    attn_flash_wmma<<<dim3(S / 128, H), blk, 0, stream>>>(Qb, Kb, Vb, Zb);

    // out = Z[S, H*E] @ proj[H*E, E], f32 out.
    gemm_bf16_wmma<bf16, false><<<dim3(S / 128, 1, 1), blk, 0, stream>>>(
        Zb, proj, nullptr, d_out, S, E, H * E, 0, 0, 0, 1.0f);
}
